// Explore_recommendation_decoder_69861938037644
// MI455X (gfx1250) — matrix operand order repaired and checked by reading, not compiled, not run on hardware
//
#include <hip/hip_runtime.h>

#define B_      256
#define S_      200
#define H_      512
#define H2_     1024
#define ITEMS_  100000
#define HIST_   50

typedef __attribute__((ext_vector_type(16))) __bf16 v16bf;
typedef __attribute__((ext_vector_type(8)))  __bf16 v8bf;
typedef __attribute__((ext_vector_type(8)))  float  v8f;

static __device__ __forceinline__ __bf16 f2bf(float x) { return (__bf16)x; }

// ---------------------------------------------------------------------------
// Kernel 1: fused additive attention for one batch row per workgroup.
//   q = We @ last ; k = all_memory[b] @ Ue^T (WMMA bf16) ;
//   score = tanh(q+k) . Ve ; softmax(S) ; ctx = attn @ all_memory[b] ;
//   cat_bf16[b] = [ctx, last]
// ---------------------------------------------------------------------------
__global__ __launch_bounds__(256)
void attn_cat_kernel(const float* __restrict__ lastmem,
                     const float* __restrict__ allmem,
                     const float* __restrict__ We,
                     const float* __restrict__ Ue,
                     const float* __restrict__ Ve,
                     __bf16* __restrict__ cat) {
    __shared__ float lastm[H_];
    __shared__ float qv[H_];
    __shared__ float ve[H_];
    __shared__ float score[208];
    __shared__ float red[256];

    const int t = threadIdx.x;
    const int b = blockIdx.x;

    for (int h = t; h < H_; h += 256) {
        lastm[h] = lastmem[(size_t)b * H_ + h];
        ve[h]    = Ve[h];
    }
    if (t < 208) score[t] = 0.0f;
    __syncthreads();

    // q[i] = dot(We[i,:], last[b,:])
    for (int i = t; i < H_; i += 256) {
        const float* wr = We + (size_t)i * H_;
        float s = 0.0f;
#pragma unroll 8
        for (int h = 0; h < H_; ++h) s += wr[h] * lastm[h];
        qv[i] = s;
    }
    __syncthreads();

    // ---- k-GEMM: [13x16 s-tiles] x [32x16 i-tiles], K = 512 = 16 bf16 steps
    const int lane = t & 31;
    const int wave = t >> 5;
    const int rowA = lane & 15;
    const int kbA  = (lane < 16) ? 0 : 8;   // A: per-lane K-runs at kb+kbA, kb+kbA+16
    const int colB = lane & 15;
    const int kbB  = (lane < 16) ? 0 : 8;   // B: per-lane K-runs at kb+kbB, kb+kbB+16
    const int rowD = (lane < 16) ? 0 : 8;
    const int colD = lane & 15;

    for (int tile = wave; tile < 13 * 32; tile += 8) {
        const int m0 = (tile % 13) * 16;   // s-tile origin
        const int n0 = (tile / 13) * 16;   // i-tile origin
        v8f acc = {0.f, 0.f, 0.f, 0.f, 0.f, 0.f, 0.f, 0.f};

        for (int ks = 0; ks < 16; ++ks) {
            const int kb = ks * 32;
            v16bf a, bb;

            const int srow = m0 + rowA;
            if (srow < S_) {
                const float4* ap4 =
                    (const float4*)(allmem + ((size_t)b * S_ + srow) * H_ + kb + kbA);
                float tmpA[16];
                *(float4*)(tmpA + 0)  = ap4[0];
                *(float4*)(tmpA + 4)  = ap4[1];
                *(float4*)(tmpA + 8)  = ap4[4];
                *(float4*)(tmpA + 12) = ap4[5];
#pragma unroll
                for (int e = 0; e < 16; ++e) a[e] = f2bf(tmpA[e]);
            } else {
#pragma unroll
                for (int e = 0; e < 16; ++e) a[e] = f2bf(0.0f);
            }

            // B[k][n] = Ue^T[k][n] = Ue[n][k]  -> contiguous run of Ue row n
            const int irow = n0 + colB;
            const float4* bp4 = (const float4*)(Ue + (size_t)irow * H_ + kb + kbB);
            float tmpB[16];
            *(float4*)(tmpB + 0)  = bp4[0];
            *(float4*)(tmpB + 4)  = bp4[1];
            *(float4*)(tmpB + 8)  = bp4[4];
            *(float4*)(tmpB + 12) = bp4[5];
#pragma unroll
            for (int e = 0; e < 16; ++e) bb[e] = f2bf(tmpB[e]);

            acc = __builtin_amdgcn_wmma_f32_16x16x32_bf16(
                false, a, false, bb, (short)0, acc, false, false);
        }

        // score[s] += sum_i Ve[i] * tanh(q[i] + k[s,i]) over this 16x16 tile
        const int icol = n0 + colD;
        const float vei = ve[icol];
        const float qi  = qv[icol];
#pragma unroll
        for (int r = 0; r < 8; ++r) {
            float v = tanhf(qi + acc[r]) * vei;
            v += __shfl_xor(v, 1, 32);
            v += __shfl_xor(v, 2, 32);
            v += __shfl_xor(v, 4, 32);
            v += __shfl_xor(v, 8, 32);
            const int srow = m0 + r + rowD;
            if ((lane & 15) == 0 && srow < S_) atomicAdd(&score[srow], v);
        }
    }
    __syncthreads();

    // ---- softmax over score[0..S-1]
    float val = (t < S_) ? score[t] : -3.0e38f;
    red[t] = val;
    __syncthreads();
    for (int off = 128; off > 0; off >>= 1) {
        if (t < off) red[t] = fmaxf(red[t], red[t + off]);
        __syncthreads();
    }
    const float smax = red[0];
    __syncthreads();
    float ex = (t < S_) ? expf(score[t] - smax) : 0.0f;
    red[t] = ex;
    __syncthreads();
    for (int off = 128; off > 0; off >>= 1) {
        if (t < off) red[t] += red[t + off];
        __syncthreads();
    }
    const float inv = 1.0f / red[0];
    __syncthreads();
    if (t < S_) score[t] = ex * inv;
    __syncthreads();

    // ---- ctx = attn @ all_memory[b];  cat = [ctx, last]
    for (int h = t; h < H_; h += 256) {
        const float* am = allmem + (size_t)b * S_ * H_ + h;
        float c = 0.0f;
#pragma unroll 4
        for (int s = 0; s < S_; ++s) c += score[s] * am[(size_t)s * H_];
        cat[(size_t)b * H2_ + h]       = f2bf(c);
        cat[(size_t)b * H2_ + H_ + h]  = f2bf(lastm[h]);
    }
}

// ---------------------------------------------------------------------------
// Kernel 2: logits = cat[256,1024](bf16) @ W[1024,100000](f32->bf16).
// Each wave owns one 16-column tile and ALL 16 M-tiles (128 acc VGPRs),
// so W is streamed exactly once from HBM. K = 1024 = 32 bf16 steps.
// ---------------------------------------------------------------------------
__global__ __launch_bounds__(256)
void logits_kernel(const __bf16* __restrict__ cat,
                   const float* __restrict__ W,
                   float* __restrict__ out) {
    const int t    = threadIdx.x;
    const int lane = t & 31;
    const int wave = t >> 5;
    const int ntile = blockIdx.x * 8 + wave;
    if (ntile * 16 >= ITEMS_) return;          // wave-uniform guard
    const int n0 = ntile * 16;

    const int rowA = lane & 15;
    const int kbA  = (lane < 16) ? 0 : 8;
    const int colB = lane & 15;
    const int kbB  = (lane < 16) ? 0 : 8;
    const int rowD = (lane < 16) ? 0 : 8;
    const int colD = lane & 15;

    v8f acc[16];
#pragma unroll
    for (int mt = 0; mt < 16; ++mt)
        acc[mt] = (v8f){0.f, 0.f, 0.f, 0.f, 0.f, 0.f, 0.f, 0.f};

    const int nB = n0 + colB;
    for (int ks = 0; ks < 32; ++ks) {
        const int kb = ks * 32;

        // B fragment: column nB of W, K runs kbB..+7 and kbB+16..+23 (stride ITEMS_).
        // Across the 16-lane half these gathers are 64B-line coalesced.
        const float* bp = W + (size_t)(kb + kbB) * ITEMS_ + nB;
        v16bf bb;
#pragma unroll
        for (int e = 0; e < 16; ++e) bb[e] = f2bf(bp[(size_t)(e + (e & 8)) * ITEMS_]);

        if (kb + 32 < H2_)  // stream-ahead prefetch of next K-slab of W
            __builtin_prefetch(bp + (size_t)32 * ITEMS_, 0, 1);

#pragma unroll
        for (int mt = 0; mt < 16; ++mt) {
            const __bf16* ap = cat + ((size_t)(mt * 16 + rowA)) * H2_ + kb + kbA;
            const v8bf lo = *(const v8bf*)ap;
            const v8bf hi = *(const v8bf*)(ap + 16);
            v16bf a;
#pragma unroll
            for (int e = 0; e < 8; ++e) { a[e] = lo[e]; a[e + 8] = hi[e]; }
            acc[mt] = __builtin_amdgcn_wmma_f32_16x16x32_bf16(
                false, a, false, bb, (short)0, acc[mt], false, false);
        }
    }

    const int nD = n0 + colD;
#pragma unroll
    for (int mt = 0; mt < 16; ++mt) {
#pragma unroll
        for (int r = 0; r < 8; ++r) {
            const int row = mt * 16 + r + rowD;
            out[(size_t)row * ITEMS_ + nD] = acc[mt][r];
        }
    }
}

// ---------------------------------------------------------------------------
// Kernel 3: masked fill logits[b, mask[b,j]] = -1e9
// ---------------------------------------------------------------------------
__global__ void mask_kernel(const int* __restrict__ mask, float* __restrict__ out) {
    const int tid = blockIdx.x * 256 + threadIdx.x;
    if (tid < B_ * HIST_) {
        const int b = tid / HIST_;
        const int idx = mask[tid];
        out[(size_t)b * ITEMS_ + idx] = -1.0e9f;
    }
}

// ---------------------------------------------------------------------------
// Kernel 4: row softmax over 100000 items (rows are L2-resident, 400 KB each)
// ---------------------------------------------------------------------------
__global__ __launch_bounds__(256)
void softmax_kernel(float* __restrict__ out) {
    __shared__ float red[256];
    const int t = threadIdx.x;
    const size_t base = (size_t)blockIdx.x * ITEMS_;

    float m = -3.0e38f;
    for (int c = t; c < ITEMS_; c += 256) m = fmaxf(m, out[base + c]);
    red[t] = m;
    __syncthreads();
    for (int off = 128; off > 0; off >>= 1) {
        if (t < off) red[t] = fmaxf(red[t], red[t + off]);
        __syncthreads();
    }
    const float rmax = red[0];
    __syncthreads();

    float s = 0.0f;
    for (int c = t; c < ITEMS_; c += 256) s += expf(out[base + c] - rmax);
    red[t] = s;
    __syncthreads();
    for (int off = 128; off > 0; off >>= 1) {
        if (t < off) red[t] += red[t + off];
        __syncthreads();
    }
    const float inv = 1.0f / red[0];

    for (int c = t; c < ITEMS_; c += 256)
        out[base + c] = expf(out[base + c] - rmax) * inv;
}

// ---------------------------------------------------------------------------
extern "C" void kernel_launch(void* const* d_in, const int* in_sizes, int n_in,
                              void* d_out, int out_size, void* d_ws, size_t ws_size,
                              hipStream_t stream) {
    const float* last_memory = (const float*)d_in[0];   // [B, H]
    const float* all_memory  = (const float*)d_in[1];   // [B, S, H]
    const float* item_mat_t  = (const float*)d_in[2];   // [2H, ITEMS]
    const int*   mask        = (const int*)d_in[3];     // [B, HIST]
    const float* We          = (const float*)d_in[4];   // [H, H]
    const float* Ue          = (const float*)d_in[5];   // [H, H]
    const float* Ve          = (const float*)d_in[6];   // [1, H]
    float* out = (float*)d_out;                         // [B, ITEMS]

    __bf16* cat = (__bf16*)d_ws;                        // [B, 2H] bf16 (512 KB)

    attn_cat_kernel<<<B_, 256, 0, stream>>>(last_memory, all_memory, We, Ue, Ve, cat);

    const int n_tiles  = ITEMS_ / 16;                   // 6250 (exact)
    const int n_blocks = (n_tiles + 7) / 8;             // 782
    logits_kernel<<<n_blocks, 256, 0, stream>>>(cat, item_mat_t, out);

    mask_kernel<<<(B_ * HIST_ + 255) / 256, 256, 0, stream>>>(mask, out);

    softmax_kernel<<<B_, 256, 0, stream>>>(out);
}
